// cGRU_35115652612130
// MI455X (gfx1250) — matrix operand order repaired and checked by reading, not compiled, not run on hardware
//
#include <hip/hip_runtime.h>
#include <hip/hip_bf16.h>

// ---------------------------------------------------------------------------
// Fused context-GRU cell for MI455X (gfx1250, wave32, WMMA bf16 path).
// Software-pipelined: global f32 loads for tile k+1 are issued before the
// WMMAs of tile k (latency hidden behind the XDL pipe), converted to bf16 and
// stored to the *other* LDS buffer after the WMMAs -> one barrier per K-step.
//   r  = sigmoid(x@W_rx + hc@W_rhc + b)      (merged A = [x|hc], K=2560)
//   u  = sigmoid(x@W_ux + hc@W_uhc + b)
//   h_cand = tanh(x@W_hx + b_hx + r*(hc@W_hhc + b_hhc))
//   new_h  = u*h + (1-u)*h_cand             -> written twice (tuple output)
// One workgroup = 128(M) x 64(N) tile, 8 waves of 32x32, 4 f32 acc sets.
// ---------------------------------------------------------------------------

typedef __attribute__((ext_vector_type(16))) __bf16 v16bf;
typedef __attribute__((ext_vector_type(8)))  __bf16 v8bf;
typedef __attribute__((ext_vector_type(8)))  float  v8f;

#define B_ROWS 32768
#define EMB    512
#define HID    1024
#define CTX    1024
#define KTOT   2560        // EMB + HID + CTX

#define BLK_M  128
#define BLK_N  64
#define BLK_K  32
#define APAD   40          // padded K-stride (elements) for A tile: 80B rows
#define BPAD   40          // padded K-stride (elements) per B column: 80B
#define BTILE  (BLK_N * BPAD)

union V16 { v16bf v; v8bf h[2]; };

__device__ __forceinline__ float fast_sigmoid(float x) {
    return 1.0f / (1.0f + __expf(-x));
}
__device__ __forceinline__ float fast_tanh(float x) {
    const float e = __expf(2.0f * x);
    return (e - 1.0f) / (e + 1.0f);
}

// In-flight tile: 16 A floats + 3x8 B floats per thread = 40 VGPRs.
struct TileRegs {
    float4 a[4];
    float4 b[3][2];
};

// Issue the global_load_b128s for K-step `kt`. The s_wait_loadcnt is deferred
// by the compiler to the first use inside store_tile(), so these overlap with
// the WMMA block issued in between.
__device__ __forceinline__ void issue_tile_loads(
    TileRegs& t,
    const float* __restrict__ x, const float* __restrict__ h,
    const float* __restrict__ c,
    const float* __restrict__ Wr_x,  const float* __restrict__ Wu_x,
    const float* __restrict__ Wh_x,
    const float* __restrict__ Wr_hc, const float* __restrict__ Wu_hc,
    const float* __restrict__ Wh_hc,
    long rowBase, int colOff, int tid, int kt)
{
    // A: thread t -> row tid>>1, 16 consecutive K at kt + (tid&1)*16.
    // Region boundaries (512, 1536) are multiples of 16 -> single source.
    {
        const int  r    = tid >> 1;
        const int  kk0  = (tid & 1) * 16;
        const long grow = rowBase + r;
        const int  kg   = kt + kk0;
        const float* src;
        if (kg < EMB)            src = x + grow * EMB + kg;
        else if (kg < EMB + HID) src = h + grow * HID + (kg - EMB);
        else                     src = c + grow * CTX + (kg - EMB - HID);
        const float4* s4 = (const float4*)src;
        t.a[0] = s4[0]; t.a[1] = s4[1]; t.a[2] = s4[2]; t.a[3] = s4[3];
    }
    // B: thread t -> k = tid>>3 (0..31), cols n0..n0+7 (coalesced 256B rows).
    {
        const bool xph  = (kt < EMB);
        const int  krow = xph ? kt : (kt - EMB);
        const float* W0 = xph ? Wr_x : Wr_hc;
        const float* W1 = xph ? Wu_x : Wu_hc;
        const float* W2 = xph ? Wh_x : Wh_hc;
        const int  k    = tid >> 3;
        const int  n0   = (tid & 7) * 8;
        const long off  = (long)(krow + k) * HID + colOff + n0;
        const float4* p0 = (const float4*)(W0 + off);
        const float4* p1 = (const float4*)(W1 + off);
        const float4* p2 = (const float4*)(W2 + off);
        t.b[0][0] = p0[0]; t.b[0][1] = p0[1];
        t.b[1][0] = p1[0]; t.b[1][1] = p1[1];
        t.b[2][0] = p2[0]; t.b[2][1] = p2[1];
    }
}

// Convert staged f32 regs to bf16 and write to one LDS buffer.
// B is transposed to column-major [col][k] so a WMMA B-fragment lane reads
// K kb0..+7 and kb0+16..+23 (two aligned 16B ds_loads).
__device__ __forceinline__ void store_tile(const TileRegs& t,
                                           __bf16* Abuf, __bf16* Bbuf, int tid)
{
    {
        const int r   = tid >> 1;
        const int kk0 = (tid & 1) * 16;
        v8bf p0, p1;
        p0[0] = (__bf16)t.a[0].x; p0[1] = (__bf16)t.a[0].y;
        p0[2] = (__bf16)t.a[0].z; p0[3] = (__bf16)t.a[0].w;
        p0[4] = (__bf16)t.a[1].x; p0[5] = (__bf16)t.a[1].y;
        p0[6] = (__bf16)t.a[1].z; p0[7] = (__bf16)t.a[1].w;
        p1[0] = (__bf16)t.a[2].x; p1[1] = (__bf16)t.a[2].y;
        p1[2] = (__bf16)t.a[2].z; p1[3] = (__bf16)t.a[2].w;
        p1[4] = (__bf16)t.a[3].x; p1[5] = (__bf16)t.a[3].y;
        p1[6] = (__bf16)t.a[3].z; p1[7] = (__bf16)t.a[3].w;
        *(v8bf*)&Abuf[r * APAD + kk0]     = p0;
        *(v8bf*)&Abuf[r * APAD + kk0 + 8] = p1;
    }
    {
        const int k  = tid >> 3;
        const int n0 = (tid & 7) * 8;
#pragma unroll
        for (int w = 0; w < 3; ++w) {
            __bf16* Bw = Bbuf + w * BTILE;
            Bw[(n0 + 0) * BPAD + k] = (__bf16)t.b[w][0].x;
            Bw[(n0 + 1) * BPAD + k] = (__bf16)t.b[w][0].y;
            Bw[(n0 + 2) * BPAD + k] = (__bf16)t.b[w][0].z;
            Bw[(n0 + 3) * BPAD + k] = (__bf16)t.b[w][0].w;
            Bw[(n0 + 4) * BPAD + k] = (__bf16)t.b[w][1].x;
            Bw[(n0 + 5) * BPAD + k] = (__bf16)t.b[w][1].y;
            Bw[(n0 + 6) * BPAD + k] = (__bf16)t.b[w][1].z;
            Bw[(n0 + 7) * BPAD + k] = (__bf16)t.b[w][1].w;
        }
    }
}

// One K=32 step: load A-frags once, then for each of the 3 staged weight
// tiles load B-frags and issue 4 WMMAs into the mapped accumulator set.
__device__ __forceinline__ void mma_step(const __bf16* Ash, const __bf16* Bsh,
                                         int waveM, int waveN,
                                         int laneLo, int laneHi,
                                         v8f (&acc0)[2][2], v8f (&acc1)[2][2],
                                         v8f (&acc2)[2][2])
{
    V16 a[2];
#pragma unroll
    for (int mi = 0; mi < 2; ++mi) {
        const int row = waveM + mi * 16 + laneLo;
        const int k0  = laneHi * 8;     // ISA 16-bit A layout: K k0..+7, k0+16..+23
        a[mi].h[0] = *(const v8bf*)&Ash[row * APAD + k0];
        a[mi].h[1] = *(const v8bf*)&Ash[row * APAD + k0 + 16];
    }
#pragma unroll
    for (int w = 0; w < 3; ++w) {
        const __bf16* Bw = Bsh + w * BTILE;
        V16 b[2];
#pragma unroll
        for (int ni = 0; ni < 2; ++ni) {
            const int col = waveN + ni * 16 + laneLo;
            const int kb0 = laneHi * 8;   // ISA 16-bit B layout: K kb0..+7, kb0+16..+23
            b[ni].h[0] = *(const v8bf*)&Bw[col * BPAD + kb0];
            b[ni].h[1] = *(const v8bf*)&Bw[col * BPAD + kb0 + 16];
        }
        v8f (&acc)[2][2] = (w == 0) ? acc0 : (w == 1) ? acc1 : acc2;
#pragma unroll
        for (int mi = 0; mi < 2; ++mi)
#pragma unroll
            for (int ni = 0; ni < 2; ++ni)
                acc[mi][ni] = __builtin_amdgcn_wmma_f32_16x16x32_bf16(
                    false, a[mi].v, false, b[ni].v,
                    (short)0, acc[mi][ni], false, false);
    }
}

__global__ void __launch_bounds__(256)
cgru_wmma_kernel(const float* __restrict__ x,
                 const float* __restrict__ h,
                 const float* __restrict__ c,
                 const float* __restrict__ W_ux,  const float* __restrict__ b_ux,
                 const float* __restrict__ W_uhc, const float* __restrict__ b_uhc,
                 const float* __restrict__ W_rx,  const float* __restrict__ b_rx,
                 const float* __restrict__ W_rhc, const float* __restrict__ b_rhc,
                 const float* __restrict__ W_hx,  const float* __restrict__ b_hx,
                 const float* __restrict__ W_hhc, const float* __restrict__ b_hhc,
                 float* __restrict__ out)
{
    __shared__ __align__(16) __bf16 Abuf[2][BLK_M * APAD];   // 2 x 10.0 KB
    __shared__ __align__(16) __bf16 Bbuf[2][3 * BTILE];      // 2 x 15.0 KB

    // col-block fastest: the 16 N-blocks sharing an A row-panel run
    // adjacently -> A re-reads + all 31 MB of weights served from 192 MB L2.
    const int cb  = blockIdx.x & 15;   // 16 blocks of 64 cols
    const int rb  = blockIdx.x >> 4;   // 256 blocks of 128 rows
    const int tid = threadIdx.x;
    const int wid    = tid >> 5;
    const int lane   = tid & 31;
    const int laneLo = lane & 15;
    const int laneHi = lane >> 4;
    const int waveM  = (wid & 3) * 32;
    const int waveN  = (wid >> 2) * 32;
    const long rowBase = (long)rb * BLK_M;
    const int  colOff  = cb * BLK_N;

    v8f accR[2][2], accU[2][2], accG1[2][2], accG2[2][2];
    const v8f vzero = {0.f, 0.f, 0.f, 0.f, 0.f, 0.f, 0.f, 0.f};
#pragma unroll
    for (int mi = 0; mi < 2; ++mi)
#pragma unroll
        for (int ni = 0; ni < 2; ++ni) {
            accR[mi][ni] = vzero; accU[mi][ni] = vzero;
            accG1[mi][ni] = vzero; accG2[mi][ni] = vzero;
        }

    // ---- pipeline prologue: tile 0 -> buffer 0 ----
    TileRegs t;
    issue_tile_loads(t, x, h, c, W_rx, W_ux, W_hx, W_rhc, W_uhc, W_hhc,
                     rowBase, colOff, tid, 0);
    store_tile(t, Abuf[0], Bbuf[0], tid);
    int cur = 0;

    // ---- phase 1: K over x (accR, accU, accG1); next tile always exists ----
    for (int kt = 0; kt < EMB; kt += BLK_K) {
        __syncthreads();   // buf[cur] written; buf[cur^1] readers done
        issue_tile_loads(t, x, h, c, W_rx, W_ux, W_hx, W_rhc, W_uhc, W_hhc,
                         rowBase, colOff, tid, kt + BLK_K);
        mma_step(Abuf[cur], Bbuf[cur], waveM, waveN, laneLo, laneHi,
                 accR, accU, accG1);
        store_tile(t, Abuf[cur ^ 1], Bbuf[cur ^ 1], tid);
        cur ^= 1;
    }
    // ---- phase 2: K over hc (accR, accU, accG2) ----
    for (int kt = EMB; kt < KTOT; kt += BLK_K) {
        __syncthreads();
        const bool more = (kt + BLK_K) < KTOT;
        if (more)
            issue_tile_loads(t, x, h, c, W_rx, W_ux, W_hx, W_rhc, W_uhc, W_hhc,
                             rowBase, colOff, tid, kt + BLK_K);
        mma_step(Abuf[cur], Bbuf[cur], waveM, waveN, laneLo, laneHi,
                 accR, accU, accG2);
        if (more)
            store_tile(t, Abuf[cur ^ 1], Bbuf[cur ^ 1], tid);
        cur ^= 1;
    }

    // ---- fused epilogue: biases + sigmoid/tanh + blend, write tuple ----
    // C/D layout: lane owns column (lane&15); VGPR r holds row r + 8*(lane>>4).
    // Outputs are streaming write-once -> non-temporal to keep L2 for A/W.
    float* out2 = out + (long)B_ROWS * HID;
#pragma unroll
    for (int ni = 0; ni < 2; ++ni) {
        const int n  = colOff + waveN + ni * 16 + laneLo;
        const float bR  = b_rx[n] + b_rhc[n];
        const float bU  = b_ux[n] + b_uhc[n];
        const float bG1 = b_hx[n];
        const float bG2 = b_hhc[n];
#pragma unroll
        for (int mi = 0; mi < 2; ++mi) {
#pragma unroll
            for (int r8 = 0; r8 < 8; ++r8) {
                const long m = rowBase + waveM + mi * 16 + laneHi * 8 + r8;
                const float rr   = fast_sigmoid(accR[mi][ni][r8] + bR);
                const float uu   = fast_sigmoid(accU[mi][ni][r8] + bU);
                const float cand = fast_tanh(accG1[mi][ni][r8] + bG1 +
                                             rr * (accG2[mi][ni][r8] + bG2));
                const float hv = h[m * HID + n];
                const float nh = uu * hv + (1.0f - uu) * cand;
                __builtin_nontemporal_store(nh, &out [m * HID + n]);
                __builtin_nontemporal_store(nh, &out2[m * HID + n]);
            }
        }
    }
}

extern "C" void kernel_launch(void* const* d_in, const int* in_sizes, int n_in,
                              void* d_out, int out_size, void* d_ws, size_t ws_size,
                              hipStream_t stream) {
    (void)in_sizes; (void)n_in; (void)out_size; (void)d_ws; (void)ws_size;
    const float* x     = (const float*)d_in[0];
    const float* h     = (const float*)d_in[1];
    const float* c     = (const float*)d_in[2];
    const float* W_ux  = (const float*)d_in[3];
    const float* b_ux  = (const float*)d_in[4];
    const float* W_uhc = (const float*)d_in[5];
    const float* b_uhc = (const float*)d_in[6];
    const float* W_rx  = (const float*)d_in[7];
    const float* b_rx  = (const float*)d_in[8];
    const float* W_rhc = (const float*)d_in[9];
    const float* b_rhc = (const float*)d_in[10];
    const float* W_hx  = (const float*)d_in[11];
    const float* b_hx  = (const float*)d_in[12];
    const float* W_hhc = (const float*)d_in[13];
    const float* b_hhc = (const float*)d_in[14];
    float* out = (float*)d_out;

    const int grid = (B_ROWS / BLK_M) * (HID / BLK_N);  // 256 * 16 = 4096
    cgru_wmma_kernel<<<grid, 256, 0, stream>>>(
        x, h, c, W_ux, b_ux, W_uhc, b_uhc, W_rx, b_rx, W_rhc, b_rhc,
        W_hx, b_hx, W_hhc, b_hhc, out);
}
